// RQVAE_25864293056553
// MI455X (gfx1250) — matrix operand order repaired and checked by reading, not compiled, not run on hardware
//
#include <hip/hip_runtime.h>

// ---------------------------------------------------------------------------
// CDNA5 (gfx1250) RQ-VAE forward:
//   bf16 WMMA GEMM pipeline (async global->LDS tile DMA, double-buffered)
//   + LDS-resident residual VQ
// ---------------------------------------------------------------------------

typedef __attribute__((ext_vector_type(16))) __bf16 v16bf;
typedef __attribute__((ext_vector_type(8)))  float  v8f;

__device__ __forceinline__ unsigned short f2bf(float f) {
  unsigned int u = __float_as_uint(f);
  u += 0x7FFFu + ((u >> 16) & 1u);   // round-to-nearest-even
  return (unsigned short)(u >> 16);
}

// Async copy of 16B (8 bf16) from global to LDS; tracked by ASYNCcnt.
__device__ __forceinline__ void async_b128_to_lds(const unsigned short* gptr,
                                                  unsigned short* lptr) {
  unsigned lds_off = (unsigned)(unsigned long long)lptr;           // addr[31:0] = LDS offset
  unsigned long long ga = (unsigned long long)gptr;
  asm volatile("global_load_async_to_lds_b128 %0, %1, off"
               :: "v"(lds_off), "v"(ga) : "memory");
}
__device__ __forceinline__ void wait_async0() {
  asm volatile("s_wait_asynccnt 0" ::: "memory");
}

// --------------------------- f32 -> bf16 convert ---------------------------
__global__ __launch_bounds__(256) void cvt_f32_bf16_x4(
    const float* __restrict__ in, unsigned short* __restrict__ out, int n4) {
  int i = blockIdx.x * 256 + threadIdx.x;
  if (i >= n4) return;
  float4 v = ((const float4*)in)[i];
  unsigned long long p =
      (unsigned long long)f2bf(v.x) |
      ((unsigned long long)f2bf(v.y) << 16) |
      ((unsigned long long)f2bf(v.z) << 32) |
      ((unsigned long long)f2bf(v.w) << 48);
  ((unsigned long long*)out)[i] = p;
}

__global__ void zero_scalar(float* p) {
  if (threadIdx.x == 0 && blockIdx.x == 0) *p = 0.f;
}

// ------------------------------- WMMA GEMM ---------------------------------
// Y[M,N] = relu?(A[M,K] @ W[N,K]^T + bias)   (A,W bf16; f32 accumulate)
// 256 threads = 8 waves arranged WM x WN; each wave owns a 32x32 output tile
// (four v_wmma_f32_16x16x32_bf16 accumulators). BM = WM*32, BN = WN*32.
// Tiles staged with GLOBAL_LOAD_ASYNC_TO_LDS_B128, double-buffered in LDS.
template <int WM, int WN, bool RELU, bool SF32, bool SBF16>
__global__ __launch_bounds__(256) void gemm_bf16_wmma(
    const unsigned short* __restrict__ A,   // M x K  bf16 row-major
    const unsigned short* __restrict__ W,   // N x K  bf16 row-major
    const float* __restrict__ bias,         // N
    unsigned short* __restrict__ Obf,       // M x N  bf16 (if SBF16)
    float* __restrict__ Of,                 // M x N  f32  (if SF32)
    int M, int N, int K) {
  constexpr int BM = WM * 32;
  constexpr int BN = WN * 32;
  constexpr int ATILE = BM * 32;            // ushorts per A tile
  constexpr int BTILE = BN * 32;            // ushorts per B tile
  __shared__ unsigned short lds_a[2 * ATILE];
  __shared__ unsigned short lds_b[2 * BTILE];

  const int tid   = threadIdx.x;
  const int lane  = tid & 31;
  const int wid   = tid >> 5;
  const int waveM = wid % WM;
  const int waveN = wid / WM;
  const int mOff  = waveM * 32;
  const int nOff  = waveN * 32;
  const int blockM = blockIdx.x * BM;
  const int blockN = blockIdx.y * BN;

  const int m_lane = lane & 15;
  const int kh     = lane >> 4;     // K-half selector per ISA fragment layout

  v8f c00 = {}, c01 = {}, c10 = {}, c11 = {};

  // ---- async tile fill: each lane moves 16B straight into LDS ----
  auto issue_tiles = [&](int k0, int buf) {
    #pragma unroll
    for (int t = 0; t < BM / 64; ++t) {          // A: BM x 32 bf16
      int idx = (t * 256 + tid) * 8;
      int row = idx >> 5, col = idx & 31;
      async_b128_to_lds(&A[(blockM + row) * K + k0 + col],
                        &lds_a[buf * ATILE + idx]);
    }
    #pragma unroll
    for (int t = 0; t < (BN * 4 + 255) / 256; ++t) {  // B: BN x 32 bf16
      int tt = t * 256 + tid;
      if (BN * 4 >= 256 || tt < BN * 4) {
        int idx = tt * 8;
        int row = idx >> 5, col = idx & 31;
        async_b128_to_lds(&W[(blockN + row) * K + k0 + col],
                          &lds_b[buf * BTILE + idx]);
      }
    }
  };

  issue_tiles(0, 0);
  wait_async0();
  __syncthreads();

  int buf = 0;
  for (int k0 = 0; k0 < K; k0 += 32) {
    const bool more = (k0 + 32) < K;
    if (more) issue_tiles(k0 + 32, buf ^ 1);     // prefetch next tile pair

    union { v16bf v; unsigned int u[8]; } fa0, fa1, fb0, fb1;
    // A fragments (16x32): lanes 0-15 -> K{0..7,16..23}, lanes 16-31 -> K{8..15,24..31}
    const unsigned short* ap0 = &lds_a[buf * ATILE + (mOff +      m_lane) * 32];
    const unsigned short* ap1 = &lds_a[buf * ATILE + (mOff + 16 + m_lane) * 32];
    #pragma unroll
    for (int i = 0; i < 8; ++i) {
      int k = (i >> 2) * 16 + kh * 8 + (i & 3) * 2;
      fa0.u[i] = *(const unsigned int*)(&ap0[k]);
      fa1.u[i] = *(const unsigned int*)(&ap1[k]);
    }
    // B fragments (32x16): VGPR j holds K = (j>>2)*16 + kh*8 + (j&3)*2 (+1), N = lane&15
    const unsigned short* bp0 = &lds_b[buf * BTILE + (nOff +      m_lane) * 32 + kh * 8];
    const unsigned short* bp1 = &lds_b[buf * BTILE + (nOff + 16 + m_lane) * 32 + kh * 8];
    #pragma unroll
    for (int j = 0; j < 8; ++j) {
      fb0.u[j] = *(const unsigned int*)(&bp0[(j >> 2) * 16 + (j & 3) * 2]);
      fb1.u[j] = *(const unsigned int*)(&bp1[(j >> 2) * 16 + (j & 3) * 2]);
    }
    c00 = __builtin_amdgcn_wmma_f32_16x16x32_bf16(false, fa0.v, false, fb0.v, (short)0, c00, false, false);
    c01 = __builtin_amdgcn_wmma_f32_16x16x32_bf16(false, fa0.v, false, fb1.v, (short)0, c01, false, false);
    c10 = __builtin_amdgcn_wmma_f32_16x16x32_bf16(false, fa1.v, false, fb0.v, (short)0, c10, false, false);
    c11 = __builtin_amdgcn_wmma_f32_16x16x32_bf16(false, fa1.v, false, fb1.v, (short)0, c11, false, false);

    if (more) {                // next tile resident AND everyone done reading `buf`
      wait_async0();
      __syncthreads();
      buf ^= 1;
    }
  }

  // Epilogue: C/D layout — lane n = lane&15, rows M = (lane>>4)*8 + r
  const int nl   = lane & 15;
  const int mb   = (lane >> 4) * 8;
  const int col0 = blockN + nOff + nl;
  const int col1 = col0 + 16;
  const float b0 = bias[col0];
  const float b1 = bias[col1];
  #pragma unroll
  for (int r2 = 0; r2 < 8; ++r2) {
    int row0 = blockM + mOff + mb + r2;
    int row1 = row0 + 16;
    float v00 = c00[r2] + b0, v01 = c01[r2] + b1;
    float v10 = c10[r2] + b0, v11 = c11[r2] + b1;
    if (RELU) {
      v00 = v00 > 0.f ? v00 : 0.f;  v01 = v01 > 0.f ? v01 : 0.f;
      v10 = v10 > 0.f ? v10 : 0.f;  v11 = v11 > 0.f ? v11 : 0.f;
    }
    if (SF32) {
      Of[row0 * N + col0] = v00;  Of[row0 * N + col1] = v01;
      Of[row1 * N + col0] = v10;  Of[row1 * N + col1] = v11;
    }
    if (SBF16) {
      Obf[row0 * N + col0] = f2bf(v00);  Obf[row0 * N + col1] = f2bf(v01);
      Obf[row1 * N + col0] = f2bf(v10);  Obf[row1 * N + col1] = f2bf(v11);
    }
  }
}

// ----------------------------- residual VQ ---------------------------------
__global__ __launch_bounds__(256) void rvq_kernel(
    const float* __restrict__ z,
    const float* __restrict__ codebooks,     // [4][256][32]
    float* __restrict__ xq_out,              // B x 32 f32 (d_out region)
    unsigned short* __restrict__ xq_bf,      // B x 32 bf16 (decoder input)
    float* __restrict__ idx_out,             // B x 4 (as float values)
    float* __restrict__ loss_out) {
  __shared__ float cb[256 * 32];
  __shared__ float red[256];
  const int tid = threadIdx.x;
  const int row = blockIdx.x * 256 + tid;

  float r[32], q[32];
  #pragma unroll
  for (int j = 0; j < 8; ++j) {
    float4 v = ((const float4*)(z + row * 32))[j];
    r[4*j] = v.x; r[4*j+1] = v.y; r[4*j+2] = v.z; r[4*j+3] = v.w;
    q[4*j] = 0.f; q[4*j+1] = 0.f; q[4*j+2] = 0.f; q[4*j+3] = 0.f;
  }
  float lsum = 0.f;
  for (int s = 0; s < 4; ++s) {
    __syncthreads();
    const float4* src = (const float4*)(codebooks + s * 256 * 32);
    float4* dst = (float4*)cb;
    #pragma unroll
    for (int j = 0; j < 8; ++j) dst[tid * 8 + j] = src[tid * 8 + j];
    __syncthreads();

    float best = 3.4e38f; int bi = 0;
    for (int c = 0; c < 256; ++c) {
      const float4* e4 = (const float4*)(cb + c * 32);
      float d = 0.f;
      #pragma unroll
      for (int j = 0; j < 8; ++j) {
        float4 e = e4[j];
        float t0 = r[4*j]   - e.x;
        float t1 = r[4*j+1] - e.y;
        float t2 = r[4*j+2] - e.z;
        float t3 = r[4*j+3] - e.w;
        d = fmaf(t0, t0, d); d = fmaf(t1, t1, d);
        d = fmaf(t2, t2, d); d = fmaf(t3, t3, d);
      }
      if (d < best) { best = d; bi = c; }   // strict '<' == argmin first-hit
    }
    lsum += best;
    const float* e = cb + bi * 32;
    #pragma unroll
    for (int k = 0; k < 32; ++k) { q[k] += e[k]; r[k] -= e[k]; }
    idx_out[row * 4 + s] = (float)bi;
  }
  #pragma unroll
  for (int k = 0; k < 32; ++k) xq_out[row * 32 + k] = q[k];
  #pragma unroll
  for (int j = 0; j < 8; ++j) {
    unsigned long long p =
        (unsigned long long)f2bf(q[4*j]) |
        ((unsigned long long)f2bf(q[4*j+1]) << 16) |
        ((unsigned long long)f2bf(q[4*j+2]) << 32) |
        ((unsigned long long)f2bf(q[4*j+3]) << 48);
    ((unsigned long long*)(xq_bf + row * 32))[j] = p;
  }
  red[tid] = lsum;
  __syncthreads();
  for (int off = 128; off > 0; off >>= 1) {
    if (tid < off) red[tid] += red[tid + off];
    __syncthreads();
  }
  if (tid == 0)
    atomicAdd(loss_out, red[0] * (1.25f / (4.0f * 65536.0f * 32.0f)));
}

// ------------------------------- launcher ----------------------------------
extern "C" void kernel_launch(void* const* d_in, const int* in_sizes, int n_in,
                              void* d_out, int out_size, void* d_ws, size_t ws_size,
                              hipStream_t stream) {
  const float* x     = (const float*)d_in[0];
  const float* ew[4] = {(const float*)d_in[1], (const float*)d_in[3],
                        (const float*)d_in[5], (const float*)d_in[7]};
  const float* eb[4] = {(const float*)d_in[2], (const float*)d_in[4],
                        (const float*)d_in[6], (const float*)d_in[8]};
  const float* dw[4] = {(const float*)d_in[9],  (const float*)d_in[11],
                        (const float*)d_in[13], (const float*)d_in[15]};
  const float* db[4] = {(const float*)d_in[10], (const float*)d_in[12],
                        (const float*)d_in[14], (const float*)d_in[16]};
  const float* cbk   = (const float*)d_in[17];
  float* out = (float*)d_out;

  // Workspace layout (bytes)
  char* ws = (char*)d_ws;
  unsigned short* bufA = (unsigned short*)(ws + 0);            // 65536x768 bf16
  unsigned short* bufB = (unsigned short*)(ws + 100663296ULL); // 65536x512 bf16
  float*          zbuf = (float*)        (ws + 167772160ULL);  // 65536x32  f32
  unsigned short* xqbf = (unsigned short*)(ws + 176160768ULL); // 65536x32  bf16
  unsigned short* wbf  = (unsigned short*)(ws + 180355072ULL); // all weights bf16
  unsigned short* ewb[4] = {wbf + 0, wbf + 393216, wbf + 524288, wbf + 557056};
  unsigned short* dwb[4] = {wbf + 561152, wbf + 565248, wbf + 598016, wbf + 729088};
  const int ewn[4] = {393216, 131072, 32768, 4096};
  const int dwn[4] = {4096, 32768, 131072, 393216};

  // Output layout (float elements)
  const int LOSS_OFF = 50331648;             // after out (65536*768)
  const int IDX_OFF  = 50331649;             // B*4 indices
  const int XQ_OFF   = 50593793;             // B*32 x_q

  for (int i = 0; i < 4; ++i) {
    cvt_f32_bf16_x4<<<(ewn[i] / 4 + 255) / 256, 256, 0, stream>>>(ew[i], ewb[i], ewn[i] / 4);
    cvt_f32_bf16_x4<<<(dwn[i] / 4 + 255) / 256, 256, 0, stream>>>(dw[i], dwb[i], dwn[i] / 4);
  }
  cvt_f32_bf16_x4<<<(50331648 / 4) / 256, 256, 0, stream>>>(x, bufA, 50331648 / 4);
  zero_scalar<<<1, 1, 0, stream>>>(out + LOSS_OFF);

  const int M = 65536;
  // ---- encoder ----
  gemm_bf16_wmma<4, 2, true,  false, true ><<<dim3(M / 128, 512 / 64), 256, 0, stream>>>(
      bufA, ewb[0], eb[0], bufB, nullptr, M, 512, 768);
  gemm_bf16_wmma<4, 2, true,  false, true ><<<dim3(M / 128, 256 / 64), 256, 0, stream>>>(
      bufB, ewb[1], eb[1], bufA, nullptr, M, 256, 512);
  gemm_bf16_wmma<4, 2, true,  false, true ><<<dim3(M / 128, 128 / 64), 256, 0, stream>>>(
      bufA, ewb[2], eb[2], bufB, nullptr, M, 128, 256);
  gemm_bf16_wmma<8, 1, false, true,  false><<<dim3(M / 256, 1), 256, 0, stream>>>(
      bufB, ewb[3], eb[3], nullptr, zbuf, M, 32, 128);
  // ---- residual VQ ----
  rvq_kernel<<<M / 256, 256, 0, stream>>>(zbuf, cbk, out + XQ_OFF, xqbf,
                                          out + IDX_OFF, out + LOSS_OFF);
  // ---- decoder ----
  gemm_bf16_wmma<4, 2, true,  false, true ><<<dim3(M / 128, 128 / 64), 256, 0, stream>>>(
      xqbf, dwb[0], db[0], bufA, nullptr, M, 128, 32);
  gemm_bf16_wmma<4, 2, true,  false, true ><<<dim3(M / 128, 256 / 64), 256, 0, stream>>>(
      bufA, dwb[1], db[1], bufB, nullptr, M, 256, 128);
  gemm_bf16_wmma<4, 2, true,  false, true ><<<dim3(M / 128, 512 / 64), 256, 0, stream>>>(
      bufB, dwb[2], db[2], bufA, nullptr, M, 512, 256);
  gemm_bf16_wmma<4, 2, false, true,  false><<<dim3(M / 128, 768 / 64), 256, 0, stream>>>(
      bufA, dwb[3], db[3], nullptr, out, M, 768, 512);
}
